// GraphSageLayer_7301444403799
// MI455X (gfx1250) — matrix operand order repaired and checked by reading, not compiled, not run on hardware
//
#include <hip/hip_runtime.h>
#include <math.h>

// ---------------------------------------------------------------------------
// CDNA5 WMMA (wave32): D(f32 16x16) = A(bf16 16x32) x B(bf16 32x16) + C
// ---------------------------------------------------------------------------
typedef __attribute__((ext_vector_type(16))) __bf16 v16bf;
typedef __attribute__((ext_vector_type(8)))  float  v8f;

union AFragU { v16bf v; unsigned short s[16]; uint4 u[2]; };
union CFragU { v8f   v; float f[8]; };
union U128   { uint4 u; unsigned short s[8]; };

// Native f32 -> bf16 convert (RTNE). Backend emits hardware cvt on gfx1250
// (and can fuse adjacent pairs into packed converts).
__device__ __forceinline__ unsigned short f2bf(float f) {
    union { __bf16 h; unsigned short u; } c;
    c.h = (__bf16)f;
    return c.u;
}

#define NNODES 4096
#define DIN    128
#define MROWS  (4 * NNODES)   // B*N = 16384

// ---------------------------------------------------------------------------
// Kernel 1: fused 3-way projection  rep = ELU(nodes @ W^T + b)
//   grid = (M/16, 24) ; block = 32 (one wave per 16x16 tile)
//   proj0/1 write rep TRANSPOSED: rep_T[b][feat][node]  (bf16) so the
//   aggregation B-fragments become contiguous b128 global loads.
//   proj2 (node_rep) goes row-major into upd cols 128..255.
// ---------------------------------------------------------------------------
__global__ void __launch_bounds__(32)
proj_kernel(const float* __restrict__ nodes,
            const float* __restrict__ W_in,  const float* __restrict__ b_in,
            const float* __restrict__ W_out, const float* __restrict__ b_out,
            const float* __restrict__ W_node,const float* __restrict__ b_node,
            unsigned short* __restrict__ in_repT,    // [B][128][N]
            unsigned short* __restrict__ out_repT,   // [B][128][N]
            unsigned short* __restrict__ upd)        // [B*N][384]
{
    const int m0   = blockIdx.x * 16;
    const int ct   = blockIdx.y;          // 0..23
    const int proj = ct >> 3;             // 0,1,2
    const int cb   = (ct & 7) * 16;       // feature-col base within 128

    const float* W;  const float* bias;
    if (proj == 0)      { W = W_in;   bias = b_in;   }
    else if (proj == 1) { W = W_out;  bias = b_out;  }
    else                { W = W_node; bias = b_node; }

    const int lane = threadIdx.x;
    const int half = lane >> 4;           // 0: K{0..15}  1: K{16..31}
    const int ml   = lane & 15;

    CFragU acc; acc.v = (v8f){0.f,0.f,0.f,0.f,0.f,0.f,0.f,0.f};

    #pragma unroll
    for (int k0 = 0; k0 < DIN; k0 += 32) {
        // ---- A fragment: nodes row m0+ml ----
        AFragU av;
        const float* ap = nodes + (size_t)(m0 + ml) * DIN + k0 + half * 16;
        float4 x0 = *(const float4*)(ap + 0);
        float4 x1 = *(const float4*)(ap + 4);
        float4 x2 = *(const float4*)(ap + 8);
        float4 x3 = *(const float4*)(ap + 12);
        av.s[0]=f2bf(x0.x); av.s[1]=f2bf(x0.y); av.s[2]=f2bf(x0.z); av.s[3]=f2bf(x0.w);
        av.s[4]=f2bf(x1.x); av.s[5]=f2bf(x1.y); av.s[6]=f2bf(x1.z); av.s[7]=f2bf(x1.w);
        av.s[8]=f2bf(x2.x); av.s[9]=f2bf(x2.y); av.s[10]=f2bf(x2.z); av.s[11]=f2bf(x2.w);
        av.s[12]=f2bf(x3.x);av.s[13]=f2bf(x3.y);av.s[14]=f2bf(x3.z); av.s[15]=f2bf(x3.w);

        // ---- B fragment: B[k][r] = W[r][k] ; lane holds out-col r = cb+ml ----
        AFragU bv;
        const float* wp = W + (size_t)(cb + ml) * DIN + k0 + half * 16;
        float4 y0 = *(const float4*)(wp + 0);
        float4 y1 = *(const float4*)(wp + 4);
        float4 y2 = *(const float4*)(wp + 8);
        float4 y3 = *(const float4*)(wp + 12);
        bv.s[0]=f2bf(y0.x); bv.s[1]=f2bf(y0.y); bv.s[2]=f2bf(y0.z); bv.s[3]=f2bf(y0.w);
        bv.s[4]=f2bf(y1.x); bv.s[5]=f2bf(y1.y); bv.s[6]=f2bf(y1.z); bv.s[7]=f2bf(y1.w);
        bv.s[8]=f2bf(y2.x); bv.s[9]=f2bf(y2.y); bv.s[10]=f2bf(y2.z); bv.s[11]=f2bf(y2.w);
        bv.s[12]=f2bf(y3.x);bv.s[13]=f2bf(y3.y);bv.s[14]=f2bf(y3.z); bv.s[15]=f2bf(y3.w);

        acc.v = __builtin_amdgcn_wmma_f32_16x16x32_bf16(
            false, av.v, false, bv.v, (short)0, acc.v, false, false);
    }

    const int col = cb + ml;
    const float bb = bias[col];
    const int rowBase = half * 8;
    const int bidx  = m0 >> 12;            // batch (N = 4096)
    const int node0 = (m0 & (NNODES - 1)) + rowBase;

    if (proj < 2) {
        // rep_T[bidx][col][node0 .. node0+7]  -> one aligned 16-B store
        U128 w;
        #pragma unroll
        for (int v = 0; v < 8; v++) {
            float x = acc.f[v] + bb;
            float e = (x > 0.0f) ? x : (__expf(x) - 1.0f);
            w.s[v] = f2bf(e);
        }
        unsigned short* dst = (proj == 0 ? in_repT : out_repT)
                            + ((size_t)bidx * 128 + col) * NNODES + node0;
        *(uint4*)dst = w.u;
    } else {
        #pragma unroll
        for (int v = 0; v < 8; v++) {
            float x = acc.f[v] + bb;
            float e = (x > 0.0f) ? x : (__expf(x) - 1.0f);
            size_t row = (size_t)(m0 + rowBase + v);
            upd[row * 384 + 128 + col] = f2bf(e);
        }
    }
}

// ---------------------------------------------------------------------------
// Kernel 2: aggregation.  transposed==0:  out[i,:] += adj[i,k] * rep[k,:]
//                         transposed==1:  out[j,:] += adj[k,j] * rep[k,:]
//   grid = (N/64, B) ; block = 256 (8 waves)
//   Workgroup tile: 64 output rows x 128 features, K-step 64.
//   - adj converted f32->bf16 ONCE at staging into LDS (only LDS user).
//   - B fragments load straight from rep_T (L2-resident), 2 x b128 per frag.
//   - Each wave: 4 M-subtiles share one B fragment -> 8 WMMAs per K-tile.
// ---------------------------------------------------------------------------
__global__ void __launch_bounds__(256)
agg_kernel(const float* __restrict__ adj,            // [B,N,N] f32
           const unsigned short* __restrict__ repT,  // [B][128][N] bf16
           unsigned short* __restrict__ upd,         // [B*N][384] bf16
           int colBase, int transposed)
{
    const int N  = NNODES;
    const int b  = blockIdx.y;
    const int r0 = blockIdx.x * 64;                  // output row tile (64 rows)
    const float* A = adj + (size_t)b * N * N;
    const unsigned short* RbT = repT + (size_t)b * 128 * N;

    // rows padded to 72 shorts (144 B = 9*16 B) -> all frag reads 16B-aligned
    __shared__ unsigned short As[64][72];            // adj tile [m][k] bf16

    const int tid  = threadIdx.x;
    const int wave = tid >> 5;
    const int lane = tid & 31;
    const int half = lane >> 4;
    const int ml   = lane & 15;
    const int nb   = wave * 16 + ml;                 // feature col this lane owns
    const int kA   = half * 16;
    const int kB   = half * 16;
    const unsigned short* Brow = RbT + (size_t)nb * N;   // rep_T row for lane

    CFragU acc[4];
    acc[0].v = (v8f){0.f,0.f,0.f,0.f,0.f,0.f,0.f,0.f};
    acc[1].v = acc[0].v; acc[2].v = acc[0].v; acc[3].v = acc[0].v;

    for (int k0 = 0; k0 < N; k0 += 64) {
        __syncthreads();
        // ---- stage adj tile (64x64) -> bf16 in LDS ----
        if (!transposed) {
            // As[m][k] = adj[r0+m][k0+k],  m:0..63, k:0..63
            int m = tid >> 2;                // 0..63
            int k = (tid & 3) * 16;          // 0,16,32,48
            const float* p = A + (size_t)(r0 + m) * N + (k0 + k);
            float4 a0 = *(const float4*)(p + 0);
            float4 a1 = *(const float4*)(p + 4);
            float4 a2 = *(const float4*)(p + 8);
            float4 a3 = *(const float4*)(p + 12);
            U128 w0, w1;
            w0.s[0]=f2bf(a0.x); w0.s[1]=f2bf(a0.y); w0.s[2]=f2bf(a0.z); w0.s[3]=f2bf(a0.w);
            w0.s[4]=f2bf(a1.x); w0.s[5]=f2bf(a1.y); w0.s[6]=f2bf(a1.z); w0.s[7]=f2bf(a1.w);
            w1.s[0]=f2bf(a2.x); w1.s[1]=f2bf(a2.y); w1.s[2]=f2bf(a2.z); w1.s[3]=f2bf(a2.w);
            w1.s[4]=f2bf(a3.x); w1.s[5]=f2bf(a3.y); w1.s[6]=f2bf(a3.z); w1.s[7]=f2bf(a3.w);
            *(uint4*)&As[m][k + 0] = w0.u;
            *(uint4*)&As[m][k + 8] = w1.u;
            if (k0 + 64 < N) __builtin_prefetch(p + 64, 0, 1);
        } else {
            // As[m][k] = adj[k0+k][r0+m]  (row-major load, transpose into LDS)
            int k = tid >> 2;                // 0..63
            int m = (tid & 3) * 16;          // 0,16,32,48
            const float* p = A + (size_t)(k0 + k) * N + (r0 + m);
            float4 a0 = *(const float4*)(p + 0);
            float4 a1 = *(const float4*)(p + 4);
            float4 a2 = *(const float4*)(p + 8);
            float4 a3 = *(const float4*)(p + 12);
            As[m + 0][k] = f2bf(a0.x);  As[m + 1][k] = f2bf(a0.y);
            As[m + 2][k] = f2bf(a0.z);  As[m + 3][k] = f2bf(a0.w);
            As[m + 4][k] = f2bf(a1.x);  As[m + 5][k] = f2bf(a1.y);
            As[m + 6][k] = f2bf(a1.z);  As[m + 7][k] = f2bf(a1.w);
            As[m + 8][k] = f2bf(a2.x);  As[m + 9][k] = f2bf(a2.y);
            As[m +10][k] = f2bf(a2.z);  As[m +11][k] = f2bf(a2.w);
            As[m +12][k] = f2bf(a3.x);  As[m +13][k] = f2bf(a3.y);
            As[m +14][k] = f2bf(a3.z);  As[m +15][k] = f2bf(a3.w);
            if (k0 + 64 < N) __builtin_prefetch(p + (size_t)64 * N, 0, 1);
        }
        __syncthreads();

        // ---- 2 K-steps; B frag direct from rep_T; 4 M-subtiles reuse it ----
        #pragma unroll
        for (int ks = 0; ks < 64; ks += 32) {
            AFragU bv;
            const unsigned short* bp = Brow + (k0 + ks + kB);
            bv.u[0] = *(const uint4*)(bp + 0);
            bv.u[1] = *(const uint4*)(bp + 8);

            #pragma unroll
            for (int ms = 0; ms < 4; ms++) {
                AFragU a;
                a.u[0] = *(const uint4*)&As[ms * 16 + ml][ks + kA + 0];
                a.u[1] = *(const uint4*)&As[ms * 16 + ml][ks + kA + 8];
                acc[ms].v = __builtin_amdgcn_wmma_f32_16x16x32_bf16(
                    false, a.v, false, bv.v, (short)0, acc[ms].v, false, false);
            }
        }
    }

    // ---- store the four accumulator subtiles into upd (bf16) ----
    const int col = colBase + wave * 16 + ml;
    const int rowBase = half * 8;
    #pragma unroll
    for (int ms = 0; ms < 4; ms++) {
        #pragma unroll
        for (int v = 0; v < 8; v++) {
            size_t row = (size_t)b * N + r0 + ms * 16 + rowBase + v;
            upd[row * 384 + col] = f2bf(acc[ms].f[v]);
        }
    }
}

// ---------------------------------------------------------------------------
// Kernel 3: out = tanh(upd @ W_upd^T + b_upd)   M=16384, Nout=128, K=384
//   grid = (M/16, 8) ; block = 32
// ---------------------------------------------------------------------------
__global__ void __launch_bounds__(32)
final_kernel(const unsigned short* __restrict__ upd,   // [M,384] bf16
             const float* __restrict__ W_upd,          // [128,384] f32
             const float* __restrict__ b_upd,
             float* __restrict__ out)                  // [M,128] f32
{
    const int m0 = blockIdx.x * 16;
    const int o0 = blockIdx.y * 16;
    const int lane = threadIdx.x;
    const int half = lane >> 4;
    const int ml   = lane & 15;

    CFragU acc; acc.v = (v8f){0.f,0.f,0.f,0.f,0.f,0.f,0.f,0.f};

    #pragma unroll
    for (int k0 = 0; k0 < 384; k0 += 32) {
        // ---- A fragment: bf16 loaded directly ----
        AFragU av;
        const unsigned short* ap = upd + (size_t)(m0 + ml) * 384 + k0 + half * 16;
        av.u[0] = *(const uint4*)(ap + 0);
        av.u[1] = *(const uint4*)(ap + 8);

        // ---- B fragment: W_upd row o = o0+ml, f32 -> bf16 ----
        AFragU bv;
        const float* wp = W_upd + (size_t)(o0 + ml) * 384 + k0 + half * 16;
        float4 y0 = *(const float4*)(wp + 0);
        float4 y1 = *(const float4*)(wp + 4);
        float4 y2 = *(const float4*)(wp + 8);
        float4 y3 = *(const float4*)(wp + 12);
        bv.s[0]=f2bf(y0.x); bv.s[1]=f2bf(y0.y); bv.s[2]=f2bf(y0.z); bv.s[3]=f2bf(y0.w);
        bv.s[4]=f2bf(y1.x); bv.s[5]=f2bf(y1.y); bv.s[6]=f2bf(y1.z); bv.s[7]=f2bf(y1.w);
        bv.s[8]=f2bf(y2.x); bv.s[9]=f2bf(y2.y); bv.s[10]=f2bf(y2.z); bv.s[11]=f2bf(y2.w);
        bv.s[12]=f2bf(y3.x);bv.s[13]=f2bf(y3.y);bv.s[14]=f2bf(y3.z); bv.s[15]=f2bf(y3.w);

        acc.v = __builtin_amdgcn_wmma_f32_16x16x32_bf16(
            false, av.v, false, bv.v, (short)0, acc.v, false, false);
    }

    const int col = o0 + ml;
    const float bb = b_upd[col];
    const int rowBase = half * 8;
    #pragma unroll
    for (int v = 0; v < 8; v++) {
        size_t row = (size_t)(m0 + rowBase + v);
        out[row * 128 + col] = tanhf(acc.f[v] + bb);
    }
}

// ---------------------------------------------------------------------------
// Launcher
// in order: nodes, adj, W_in, b_in, W_out, b_out, W_node, b_node, W_upd, b_upd
// ws (bf16 units): in_repT[B*128*N] | out_repT[B*128*N] | upd[M*384]  (20 MB)
// ---------------------------------------------------------------------------
extern "C" void kernel_launch(void* const* d_in, const int* in_sizes, int n_in,
                              void* d_out, int out_size, void* d_ws, size_t ws_size,
                              hipStream_t stream) {
    const float* nodes  = (const float*)d_in[0];
    const float* adj    = (const float*)d_in[1];
    const float* W_in   = (const float*)d_in[2];
    const float* b_in   = (const float*)d_in[3];
    const float* W_out  = (const float*)d_in[4];
    const float* b_out  = (const float*)d_in[5];
    const float* W_node = (const float*)d_in[6];
    const float* b_node = (const float*)d_in[7];
    const float* W_upd  = (const float*)d_in[8];
    const float* b_upd  = (const float*)d_in[9];
    float* out = (float*)d_out;

    unsigned short* ws       = (unsigned short*)d_ws;
    unsigned short* in_repT  = ws;                                // [B][128][N]
    unsigned short* out_repT = ws + (size_t)MROWS * 128;          // [B][128][N]
    unsigned short* upd      = ws + (size_t)MROWS * 256;          // [M][384]

    // 1) fused projections + ELU (rep outputs stored transposed)
    proj_kernel<<<dim3(MROWS / 16, 24), dim3(32), 0, stream>>>(
        nodes, W_in, b_in, W_out, b_out, W_node, b_node, in_repT, out_repT, upd);

    // 2) in_agg = adj @ in_rep        -> upd cols [0,128)
    agg_kernel<<<dim3(NNODES / 64, 4), dim3(256), 0, stream>>>(
        adj, in_repT, upd, 0, 0);

    // 3) out_agg = adj^T @ out_rep    -> upd cols [256,384)
    agg_kernel<<<dim3(NNODES / 64, 4), dim3(256), 0, stream>>>(
        adj, out_repT, upd, 256, 1);

    // 4) out = tanh(upd @ W_upd^T + b_upd)
    final_kernel<<<dim3(MROWS / 16, 8), dim3(32), 0, stream>>>(
        upd, W_upd, b_upd, out);
}
